// Encoder_19413252178505
// MI455X (gfx1250) — matrix operand order repaired and checked by reading, not compiled, not run on hardware
//
#include <hip/hip_runtime.h>

// ---------------------------------------------------------------------------
// Transformer encoder forward for gfx1250 (MI455X), WMMA f16 path.
// B=2, S=2048, D=1024, H=16, DK=64, DFF=4096, L=6.
// ---------------------------------------------------------------------------

typedef __attribute__((ext_vector_type(16))) _Float16 v16h;
typedef __attribute__((ext_vector_type(8)))  float    v8f;

// ---- WMMA fragment helpers (ISA 7.12.2 layouts, wave32) -------------------

// A-matrix 16x32 f16: lanes 0-15 -> M=lane, halves {K0..7, K16..23};
//                     lanes 16-31 -> M=lane-16, halves {K8..15, K24..31}.
__device__ __forceinline__ v16h ld_fragA(const _Float16* t, int ldk, int koff) {
  const int lane = threadIdx.x & 31;
  const int row  = lane & 15;
  const int k0   = (lane >> 4) << 3;          // 0 or 8
  const _Float16* p = t + row * ldk + koff;
  v16h r;
#pragma unroll
  for (int j = 0; j < 8; ++j) { r[j] = p[k0 + j]; r[8 + j] = p[k0 + 16 + j]; }
  return r;
}

// B-matrix 32x16 f16, sourced from an LDS tile holding B^T row-major
// (tile[n][k], k contiguous): lanes 0-15 -> N=lane, K{0..7,16..23}; lanes 16-31 -> K{8..15,24..31}.
__device__ __forceinline__ v16h ld_fragB(const _Float16* t, int ldk, int koff) {
  const int lane = threadIdx.x & 31;
  const int n    = lane & 15;
  const int k0   = (lane >> 4) << 3;          // 0 or 8
  const _Float16* p = t + n * ldk + koff + k0;
  v16h r;
#pragma unroll
  for (int j = 0; j < 8; ++j) { r[j] = p[j]; r[8 + j] = p[16 + j]; }
  return r;
}

__device__ __forceinline__ v8f wmma16(v16h a, v16h b, v8f c) {
  return __builtin_amdgcn_wmma_f32_16x16x32_f16(false, a, false, b, (short)0, c,
                                                false, false);
}

// ---- embedding ------------------------------------------------------------

__global__ void embed_kernel(const int* __restrict__ x, const float* __restrict__ emb,
                             const float* __restrict__ pos, float* __restrict__ h,
                             int S, int D, int total) {
  int i = blockIdx.x * 256 + threadIdx.x;
  if (i >= total) return;
  int d = i & (D - 1);
  int t = i / D;          // token = b*S + s
  int s = t & (S - 1);
  h[i] = emb[(size_t)x[t] * D + d] + pos[(size_t)s * D + d];
}

// ---- f32 -> f16 convert ---------------------------------------------------

__global__ void cvt_kernel(const float* __restrict__ in, _Float16* __restrict__ out,
                           int n) {
  int i = blockIdx.x * 256 + threadIdx.x;
  if (i < n) out[i] = (_Float16)in[i];
}

// ---- generic WMMA GEMM:  C[M,N] = act(A[M,K] * W[N,K]^T + bias + res) -----
// Block = 256 threads (8 waves). Block tile 128x64; waves arranged 4x2, each
// wave owns a 32x32 macro-tile = 2x2 WMMA tiles (4 WMMAs per K-step, fragments
// reused twice each).

template <bool BIAS, bool RELU, bool RES, bool OUT32, bool OUT16>
__global__ void __launch_bounds__(256)
gemm_kernel(const _Float16* __restrict__ A, const _Float16* __restrict__ W,
            const float* __restrict__ bias, const float* __restrict__ res,
            float* __restrict__ C32, _Float16* __restrict__ C16,
            int M, int N, int K) {
  __shared__ _Float16 As[128 * 32];   // 8 KB
  __shared__ _Float16 Ws[64 * 32];    // 4 KB
  const int tid  = threadIdx.x;
  const int wave = tid >> 5;
  const int lane = tid & 31;
  const int bm = blockIdx.y * 128, bn = blockIdx.x * 64;
  const int tm = (wave >> 1) * 32;    // 0,32,64,96
  const int tn = (wave & 1) * 32;     // 0,32

  v8f acc00 = {}, acc01 = {}, acc10 = {}, acc11 = {};

  // Staging geometry: 8-half (16B) chunks.
  const int ra0 = tid >> 2;              // 0..63
  const int ra1 = ra0 + 64;              // 64..127
  const int ca  = (tid & 3) * 8;         // 0/8/16/24 halves

  for (int k0 = 0; k0 < K; k0 += 32) {
    // Stage A tile 128x32 (two b128 chunks per thread).
    *(float4*)(&As[ra0 * 32 + ca]) =
        *(const float4*)(A + (size_t)(bm + ra0) * K + k0 + ca);
    *(float4*)(&As[ra1 * 32 + ca]) =
        *(const float4*)(A + (size_t)(bm + ra1) * K + k0 + ca);
    // Stage W tile 64x32 (one b128 chunk per thread).
    *(float4*)(&Ws[ra0 * 32 + ca]) =
        *(const float4*)(W + (size_t)(bn + ra0) * K + k0 + ca);
    // Prefetch next K tile (global_prefetch_b8).
    if (k0 + 32 < K) {
      __builtin_prefetch(A + (size_t)(bm + ra0) * K + k0 + 32 + ca, 0, 1);
      __builtin_prefetch(A + (size_t)(bm + ra1) * K + k0 + 32 + ca, 0, 1);
      __builtin_prefetch(W + (size_t)(bn + ra0) * K + k0 + 32 + ca, 0, 1);
    }
    __syncthreads();
    v16h a0 = ld_fragA(As + (tm + 0)  * 32, 32, 0);
    v16h a1 = ld_fragA(As + (tm + 16) * 32, 32, 0);
    v16h b0 = ld_fragB(Ws + (tn + 0)  * 32, 32, 0);
    v16h b1 = ld_fragB(Ws + (tn + 16) * 32, 32, 0);
    acc00 = wmma16(a0, b0, acc00);
    acc01 = wmma16(a0, b1, acc01);
    acc10 = wmma16(a1, b0, acc10);
    acc11 = wmma16(a1, b1, acc11);
    __syncthreads();
  }

  // Epilogue: C layout row = i + 8*(lane>=16), col = lane&15 per 16x16 tile.
  const int colBase = lane & 15;
  const int rowHalf = (lane >> 4) << 3;
#pragma unroll
  for (int st = 0; st < 4; ++st) {
    const v8f& acc = (st == 0) ? acc00 : (st == 1) ? acc01
                   : (st == 2) ? acc10 : acc11;
    const int smr = (st >> 1) * 16;     // sub-tile row offset
    const int smc = (st & 1) * 16;      // sub-tile col offset
    const int cn = bn + tn + smc + colBase;
    const float bv = BIAS ? bias[cn] : 0.0f;
#pragma unroll
    for (int i = 0; i < 8; ++i) {
      const int cm = bm + tm + smr + i + rowHalf;
      float v = acc[i] + bv;
      if (RES)  v += res[(size_t)cm * N + cn];
      if (RELU) v = fmaxf(v, 0.0f);
      if (OUT32) C32[(size_t)cm * N + cn] = v;
      if (OUT16) C16[(size_t)cm * N + cn] = (_Float16)v;
    }
  }
}

// ---- flash attention (per b,h; 4 waves/block, 16 q-rows per wave) ---------

__global__ void __launch_bounds__(128)
attn_kernel(const _Float16* __restrict__ Q, const _Float16* __restrict__ Kg,
            const _Float16* __restrict__ Vg, _Float16* __restrict__ ctx,
            int S, int H) {
  const int wave = threadIdx.x >> 5, lane = threadIdx.x & 31;
  const int b = blockIdx.z, h = blockIdx.y;
  const int qw = blockIdx.x * 64 + wave * 16;

  __shared__ _Float16 Qs[4][16 * 64];
  __shared__ _Float16 Ks[32 * 64];   // [key][d]  (= B^T for energy)
  __shared__ _Float16 Vs[64 * 32];   // [d][key]  (= B^T for ctx)
  __shared__ _Float16 Ps[4][16 * 32];

  // Stage this wave's Q tile.
  for (int idx = lane; idx < 128; idx += 32) {
    int r = idx >> 3, c = (idx & 7) * 8;
    size_t tok = (size_t)(b * S + qw + r);
    *(float4*)(&Qs[wave][r * 64 + c]) =
        *(const float4*)(Q + (tok * H + h) * 64 + c);
  }
  __syncthreads();
  v16h qf0 = ld_fragA(&Qs[wave][0], 64, 0);
  v16h qf1 = ld_fragA(&Qs[wave][0], 64, 32);

  float mrow[8], lrow[8];
  v8f cacc[4];
#pragma unroll
  for (int i = 0; i < 8; ++i) { mrow[i] = -1e30f; lrow[i] = 0.0f; }
#pragma unroll
  for (int t = 0; t < 4; ++t) cacc[t] = (v8f){};

  const float scale = 0.125f;  // 1/sqrt(64)

  for (int kt = 0; kt < S; kt += 32) {
    // Stage K tile 32x64 (b128 chunks).
    for (int idx = threadIdx.x; idx < 256; idx += 128) {
      int r = idx >> 3, c = (idx & 7) * 8;
      size_t tok = (size_t)(b * S + kt + r);
      *(float4*)(&Ks[r * 64 + c]) =
          *(const float4*)(Kg + (tok * H + h) * 64 + c);
    }
    // Stage V tile transposed: Vs[d][key].
    for (int idx = threadIdx.x; idx < 2048; idx += 128) {
      int key = idx >> 6, d = idx & 63;
      size_t tok = (size_t)(b * S + kt + key);
      Vs[d * 32 + key] = Vg[(tok * H + h) * 64 + d];
    }
    __syncthreads();

    // energy [16 x 32] = Q[16x64] * K^T
    v8f e0 = {}, e1 = {};
    e0 = wmma16(qf0, ld_fragB(&Ks[0], 64, 0), e0);
    e0 = wmma16(qf1, ld_fragB(&Ks[0], 64, 32), e0);
    e1 = wmma16(qf0, ld_fragB(&Ks[16 * 64], 64, 0), e1);
    e1 = wmma16(qf1, ld_fragB(&Ks[16 * 64], 64, 32), e1);

    // Online softmax per row (row = i + 8*(lane>=16); cols spread over 16 lanes).
#pragma unroll
    for (int i = 0; i < 8; ++i) {
      float a0 = e0[i] * scale, a1 = e1[i] * scale;
      float ml = fmaxf(a0, a1);
#pragma unroll
      for (int m = 8; m >= 1; m >>= 1) ml = fmaxf(ml, __shfl_xor(ml, m, 32));
      float mnew  = fmaxf(mrow[i], ml);
      float alpha = __expf(mrow[i] - mnew);
      float p0 = __expf(a0 - mnew), p1 = __expf(a1 - mnew);
      float ps = p0 + p1;
#pragma unroll
      for (int m = 8; m >= 1; m >>= 1) ps += __shfl_xor(ps, m, 32);
      lrow[i] = lrow[i] * alpha + ps;
      mrow[i] = mnew;
#pragma unroll
      for (int t = 0; t < 4; ++t) cacc[t][i] *= alpha;
      int r = i + ((lane >> 4) << 3), cc = lane & 15;
      Ps[wave][r * 32 + cc]      = (_Float16)p0;
      Ps[wave][r * 32 + 16 + cc] = (_Float16)p1;
    }
    __syncthreads();

    // ctx[16 x 64] += P[16x32] * V[32x64]
    v16h pf = ld_fragA(&Ps[wave][0], 32, 0);
#pragma unroll
    for (int t = 0; t < 4; ++t)
      cacc[t] = wmma16(pf, ld_fragB(&Vs[t * 16 * 32], 32, 0), cacc[t]);
    __syncthreads();
  }

  // Write ctx as f16 into [B,S,D] with D index = h*64 + d.
#pragma unroll
  for (int i = 0; i < 8; ++i) {
    int r = i + ((lane >> 4) << 3);
    float inv = 1.0f / lrow[i];
    size_t tok = (size_t)(b * S + qw + r);
#pragma unroll
    for (int t = 0; t < 4; ++t) {
      int d = h * 64 + t * 16 + (lane & 15);
      ctx[tok * 1024 + d] = (_Float16)(cacc[t][i] * inv);
    }
  }
}

// ---- LayerNorm over last dim (block per row) ------------------------------

__global__ void __launch_bounds__(256)
ln_kernel(const float* __restrict__ in, const float* __restrict__ g,
          const float* __restrict__ bta, float* __restrict__ out32, int D) {
  const int row = blockIdx.x;
  const float* x = in + (size_t)row * D;
  float s = 0.f, ss = 0.f;
  for (int i = threadIdx.x; i < D; i += 256) {
    float v = x[i]; s += v; ss += v * v;
  }
#pragma unroll
  for (int m = 16; m >= 1; m >>= 1) {
    s += __shfl_xor(s, m, 32);
    ss += __shfl_xor(ss, m, 32);
  }
  __shared__ float rs[8], rss[8];
  const int wave = threadIdx.x >> 5;
  if ((threadIdx.x & 31) == 0) { rs[wave] = s; rss[wave] = ss; }
  __syncthreads();
  s = 0.f; ss = 0.f;
#pragma unroll
  for (int w = 0; w < 8; ++w) { s += rs[w]; ss += rss[w]; }
  const float mu   = s / D;
  const float var  = ss / D - mu * mu;
  const float rstd = rsqrtf(var + 1e-5f);
  for (int i = threadIdx.x; i < D; i += 256) {
    out32[(size_t)row * D + i] = (x[i] - mu) * rstd * g[i] + bta[i];
  }
}

// ---------------------------------------------------------------------------

extern "C" void kernel_launch(void* const* d_in, const int* in_sizes, int n_in,
                              void* d_out, int out_size, void* d_ws, size_t ws_size,
                              hipStream_t stream) {
  (void)in_sizes; (void)n_in; (void)out_size; (void)ws_size;
  const int B = 2, S = 2048, D = 1024, H = 16, DK = 64, DFF = 4096, L = 6;
  const int T = B * S;  // 4096 tokens

  const int*   x    = (const int*)d_in[0];
  const float* emb  = (const float*)d_in[1];
  const float* pos  = (const float*)d_in[2];
  const float* wq   = (const float*)d_in[3];
  const float* wk   = (const float*)d_in[4];
  const float* wv   = (const float*)d_in[5];
  const float* wo   = (const float*)d_in[6];
  const float* bo   = (const float*)d_in[7];
  const float* w1   = (const float*)d_in[8];
  const float* b1   = (const float*)d_in[9];
  const float* w2   = (const float*)d_in[10];
  const float* b2   = (const float*)d_in[11];
  const float* ln1g = (const float*)d_in[12];
  const float* ln1b = (const float*)d_in[13];
  const float* ln2g = (const float*)d_in[14];
  const float* ln2b = (const float*)d_in[15];

  char* ws = (char*)d_ws;
  size_t off = 0;
  auto alloc = [&](size_t bytes) -> void* {
    void* p = ws + off;
    off += (bytes + 255) & ~(size_t)255;
    return p;
  };
  float*     h    = (float*)alloc((size_t)T * D * 4);
  float*     x1   = (float*)alloc((size_t)T * D * 4);
  float*     tmp  = (float*)alloc((size_t)T * D * 4);
  _Float16*  h16  = (_Float16*)alloc((size_t)T * D * 2);
  _Float16*  q16  = (_Float16*)alloc((size_t)T * D * 2);
  _Float16*  k16  = (_Float16*)alloc((size_t)T * D * 2);
  _Float16*  v16  = (_Float16*)alloc((size_t)T * D * 2);
  _Float16*  c16  = (_Float16*)alloc((size_t)T * D * 2);
  _Float16*  x116 = (_Float16*)alloc((size_t)T * D * 2);
  _Float16*  ff16 = (_Float16*)alloc((size_t)T * DFF * 2);
  _Float16*  wq16 = (_Float16*)alloc((size_t)DK * DK * 2);
  _Float16*  wk16 = (_Float16*)alloc((size_t)DK * DK * 2);
  _Float16*  wv16 = (_Float16*)alloc((size_t)DK * DK * 2);
  _Float16*  wo16 = (_Float16*)alloc((size_t)D * D * 2);
  _Float16*  w116 = (_Float16*)alloc((size_t)DFF * D * 2);
  _Float16*  w216 = (_Float16*)alloc((size_t)D * DFF * 2);

  auto cvt = [&](const float* src, _Float16* dst, int n) {
    cvt_kernel<<<(n + 255) / 256, 256, 0, stream>>>(src, dst, n);
  };

  // Embedding.
  {
    int total = T * D;
    embed_kernel<<<(total + 255) / 256, 256, 0, stream>>>(x, emb, pos, h, S, D, total);
  }

  for (int i = 0; i < L; ++i) {
    // Per-layer weight conversion to f16.
    cvt(wq + (size_t)i * DK * DK, wq16, DK * DK);
    cvt(wk + (size_t)i * DK * DK, wk16, DK * DK);
    cvt(wv + (size_t)i * DK * DK, wv16, DK * DK);
    cvt(wo + (size_t)i * D * D,   wo16, D * D);
    cvt(w1 + (size_t)i * DFF * D, w116, DFF * D);
    cvt(w2 + (size_t)i * D * DFF, w216, D * DFF);
    cvt(h, h16, T * D);

    // QKV projections: [T*H, DK] x [DK, DK]^T (per-head shared weights).
    {
      dim3 g(DK / 64, (T * H) / 128);
      gemm_kernel<false, false, false, false, true><<<g, 256, 0, stream>>>(
          h16, wq16, nullptr, nullptr, nullptr, q16, T * H, DK, DK);
      gemm_kernel<false, false, false, false, true><<<g, 256, 0, stream>>>(
          h16, wk16, nullptr, nullptr, nullptr, k16, T * H, DK, DK);
      gemm_kernel<false, false, false, false, true><<<g, 256, 0, stream>>>(
          h16, wv16, nullptr, nullptr, nullptr, v16, T * H, DK, DK);
    }

    // Attention -> ctx (f16, [B,S,D]).
    {
      dim3 g(S / 64, H, B);
      attn_kernel<<<g, 128, 0, stream>>>(q16, k16, v16, c16, S, H);
    }

    // attn_out + h:  tmp = ctx @ wo^T + bo + h.
    {
      dim3 g(D / 64, T / 128);
      gemm_kernel<true, false, true, true, false><<<g, 256, 0, stream>>>(
          c16, wo16, bo + (size_t)i * D, h, tmp, nullptr, T, D, D);
    }
    // x1 = LN1(tmp); x116 = f16(x1).
    ln_kernel<<<T, 256, 0, stream>>>(tmp, ln1g + (size_t)i * D, ln1b + (size_t)i * D,
                                     x1, D);
    cvt(x1, x116, T * D);

    // ff = relu(x1 @ w1^T + b1)  (f16).
    {
      dim3 g(DFF / 64, T / 128);
      gemm_kernel<true, true, false, false, true><<<g, 256, 0, stream>>>(
          x116, w116, b1 + (size_t)i * DFF, nullptr, nullptr, ff16, T, DFF, D);
    }
    // tmp = ff @ w2^T + b2 + x1.
    {
      dim3 g(D / 64, T / 128);
      gemm_kernel<true, false, true, true, false><<<g, 256, 0, stream>>>(
          ff16, w216, b2 + (size_t)i * D, x1, tmp, nullptr, T, D, DFF);
    }
    // h = LN2(tmp); last layer writes straight to d_out.
    float* dst = (i == L - 1) ? (float*)d_out : h;
    ln_kernel<<<T, 256, 0, stream>>>(tmp, ln2g + (size_t)i * D, ln2b + (size_t)i * D,
                                     dst, D);
  }
}
